// GCN_15264313770212
// MI455X (gfx1250) — matrix operand order repaired and checked by reading, not compiled, not run on hardware
//
#include <hip/hip_runtime.h>
#include <hip/hip_bf16.h>

typedef __attribute__((ext_vector_type(16))) __bf16 v16bf;
typedef __attribute__((ext_vector_type(8)))  float  v8f;

#define N_NODES 100000
#define N_EDGES 3200000
#define N_FEAT  512
#define N_DIMS  256
#define N_CLSS  10

// ---------------------------------------------------------------------------
// Normalization pipeline
// ---------------------------------------------------------------------------
__global__ void k_init_deg(float* __restrict__ deg, int n) {
    int i = blockIdx.x * blockDim.x + threadIdx.x;
    if (i < n) deg[i] = 1.0f;                     // self-loop weight
}

__global__ void k_accum_deg(const int* __restrict__ ei, const float* __restrict__ w,
                            float* __restrict__ deg, int E) {
    int e = blockIdx.x * blockDim.x + threadIdx.x;
    if (e < E) atomicAdd(&deg[ei[E + e]], w[e]);  // col = ei[1][e]
}

__global__ void k_dinv(const float* __restrict__ deg, float* __restrict__ dinv, int n) {
    int i = blockIdx.x * blockDim.x + threadIdx.x;
    if (i < n) { float d = deg[i]; dinv[i] = d > 0.0f ? rsqrtf(d) : 0.0f; }
}

__global__ void k_norm(const int* __restrict__ ei, const float* __restrict__ w,
                       const float* __restrict__ dinv, float* __restrict__ nrm, int E) {
    int e = blockIdx.x * blockDim.x + threadIdx.x;
    if (e < E) nrm[e] = dinv[ei[e]] * w[e] * dinv[ei[E + e]];
}

// ---------------------------------------------------------------------------
// Pack f32 [M,K] row-major into CDNA5 WMMA 16-bit A-fragment layout (bf16):
//   [M/16][K/32][lane 0..31][16 elems]   (ISA 7.12.2, 16-bit A 16x32)
// lane: m = tile*16 + (lane&15), khalf = lane>>4
// elem j: v=j>>1, k = (v<4 ? 2v : 16+2(v-4)) + 8*khalf + (j&1)
// Optionally fuses  val = relu(val + bias[k])  (bias over the K/feature dim),
// eliminating a separate full read/write pass over the activations.
// ---------------------------------------------------------------------------
__global__ void k_pack_a(const float* __restrict__ X, __bf16* __restrict__ out,
                         int M, int K, const float* __restrict__ bias) {
    int gid = blockIdx.x * blockDim.x + threadIdx.x;
    int kb32 = K >> 5;
    int total = (M >> 4) * kb32 * 32;
    if (gid >= total) return;
    int lane = gid & 31;
    int rem  = gid >> 5;
    int kb   = rem % kb32;
    int t    = rem / kb32;
    int m     = t * 16 + (lane & 15);
    int khalf = lane >> 4;
    const float* xr = X + (size_t)m * K + kb * 32;
    const float* br = bias ? bias + kb * 32 : nullptr;
    v16bf frag;
#pragma unroll
    for (int j = 0; j < 16; ++j) {
        int v = j >> 1;
        int k = (v < 4 ? 2 * v : 16 + 2 * (v - 4)) + 8 * khalf + (j & 1);
        float val = xr[k];
        if (br) val = fmaxf(val + br[k], 0.0f);   // fused bias + ReLU
        frag[j] = (__bf16)val;
    }
    *(v16bf*)(out + (size_t)gid * 16) = frag;
}

// ---------------------------------------------------------------------------
// Pack f32 W [K,N] row-major into WMMA 16-bit B-fragment layout (bf16):
//   [K/32][Npad/16][lane 0..31][16 elems]
// lane: n = nt*16 + (lane&15), khalf = lane>>4; elem j: k = kb*32 + (j<8 ? 8*khalf + j : 16 + 8*khalf + (j-8))
// Columns n >= N are zero-padded (classifier N=10 -> 16).
// ---------------------------------------------------------------------------
__global__ void k_pack_b(const float* __restrict__ W, __bf16* __restrict__ out,
                         int K, int N, int Npad) {
    int gid = blockIdx.x * blockDim.x + threadIdx.x;
    int ntc = Npad >> 4;
    int total = (K >> 5) * ntc * 32;
    if (gid >= total) return;
    int lane = gid & 31;
    int rem  = gid >> 5;
    int nt   = rem % ntc;
    int kb   = rem / ntc;
    int n     = nt * 16 + (lane & 15);
    int khalf = lane >> 4;
    v16bf frag;
#pragma unroll
    for (int j = 0; j < 16; ++j) {
        int k = kb * 32 + (j < 8 ? 8 * khalf + j : 16 + 8 * khalf + (j - 8));
        frag[j] = (__bf16)((n < N) ? W[(size_t)k * N + n] : 0.0f);
    }
    *(v16bf*)(out + (size_t)gid * 16) = frag;
}

// ---------------------------------------------------------------------------
// WMMA GEMM: C[M,N] (f32) = Apacked x Bpacked (+ optional bias on store)
// blockDim = 256 (8 waves). Each block: 8 row tiles x (NT*16) cols.
// The block's whole B slice is staged in LDS once (ds_load_b128 thereafter);
// A streams from global with a one-step register double buffer so next-kb
// loads are in flight during the current WMMAs.
// ---------------------------------------------------------------------------
template <int NT>
__global__ void k_gemm_wmma(const __bf16* __restrict__ Ap,
                            const __bf16* __restrict__ Bp,
                            float* __restrict__ C,
                            const float* __restrict__ bias,  // may be nullptr
                            int Mtiles, int Kb, int NtilesTotal,
                            int ldC, int nstore) {
    extern __shared__ char smem[];
    v16bf* shB = (v16bf*)smem;                   // [Kb][NT][32] fragments

    int tid = threadIdx.x;
    int nt0 = blockIdx.y * NT;

    // Stage B slice for this block's column group into LDS (read global once).
    int nfrag = Kb * NT * 32;
    for (int f = tid; f < nfrag; f += blockDim.x) {
        int kb = f / (NT * 32);
        int r  = f - kb * (NT * 32);
        int nt = r >> 5;
        int ln = r & 31;
        shB[f] = ((const v16bf*)Bp)[((size_t)kb * NtilesTotal + nt0 + nt) * 32 + ln];
    }
    __syncthreads();

    int wave = tid >> 5;
    int lane = tid & 31;
    int tile = blockIdx.x * (blockDim.x >> 5) + wave;
    if (tile >= Mtiles) return;                  // wave-uniform, after barrier

    v8f acc[NT] = {};
    const v16bf* Abase = (const v16bf*)Ap + (size_t)tile * Kb * 32 + lane;
    const v16bf* Bl    = shB + lane;

    v16bf a_cur = Abase[0];
    for (int kb = 0; kb < Kb - 1; ++kb) {
        v16bf a_nxt = Abase[(size_t)(kb + 1) * 32];   // in flight during WMMAs
#pragma unroll
        for (int i = 0; i < NT; ++i) {
            v16bf b = Bl[(kb * NT + i) * 32];
            acc[i] = __builtin_amdgcn_wmma_f32_16x16x32_bf16(
                false, a_cur, false, b, (short)0, acc[i], false, false);
        }
        a_cur = a_nxt;
    }
#pragma unroll
    for (int i = 0; i < NT; ++i) {               // epilogue kb = Kb-1
        v16bf b = Bl[((Kb - 1) * NT + i) * 32];
        acc[i] = __builtin_amdgcn_wmma_f32_16x16x32_bf16(
            false, a_cur, false, b, (short)0, acc[i], false, false);
    }

    int rbase = tile * 16 + ((lane >> 4) << 3);  // C layout: M = 8*(lane>=16)+v
    int cbase = lane & 15;                       //           N = lane&15
#pragma unroll
    for (int i = 0; i < NT; ++i) {
        int col = (nt0 + i) * 16 + cbase;
        if (col < nstore) {
            float bv = bias ? bias[col] : 0.0f;
#pragma unroll
            for (int v = 0; v < 8; ++v)
                C[(size_t)(rbase + v) * ldC + col] = acc[i][v] + bv;
        }
    }
}

// ---------------------------------------------------------------------------
// Aggregation: out = D^-1/2 (A + I) D^-1/2 @ h   (feature dim = 256)
// ---------------------------------------------------------------------------
__global__ void k_agg_init(const float* __restrict__ h, const float* __restrict__ dinv,
                           float* __restrict__ out) {
    // self-loop term: out[i] = dinv[i]^2 * h[i]; one thread per (node, 4 feats)
    int gid = blockIdx.x * blockDim.x + threadIdx.x;
    if (gid >= N_NODES * 64) return;
    int n  = gid >> 6;
    int f4 = gid & 63;
    float s = dinv[n] * dinv[n];
    const float4 hv = *(const float4*)(h + (size_t)n * N_DIMS + f4 * 4);
    float4 o = make_float4(s * hv.x, s * hv.y, s * hv.z, s * hv.w);
    *(float4*)(out + (size_t)n * N_DIMS + f4 * 4) = o;
}

__global__ void k_agg_edges(const int* __restrict__ ei, const float* __restrict__ nrm,
                            const float* __restrict__ h, float* __restrict__ out) {
    // 64 threads per edge, float4 gather + 4 atomic f32 scatter-adds (L2 resident)
    long long gid = (long long)blockIdx.x * blockDim.x + threadIdx.x;
    if (gid >= (long long)N_EDGES * 64) return;
    int e  = (int)(gid >> 6);
    int f4 = (int)(gid & 63);
    int r  = ei[e];
    int c  = ei[N_EDGES + e];
    float s = nrm[e];
    const float4 hv = *(const float4*)(h + (size_t)r * N_DIMS + f4 * 4);
    float* o = out + (size_t)c * N_DIMS + f4 * 4;
    atomicAdd(o + 0, s * hv.x);
    atomicAdd(o + 1, s * hv.y);
    atomicAdd(o + 2, s * hv.z);
    atomicAdd(o + 3, s * hv.w);
}

// ---------------------------------------------------------------------------
extern "C" void kernel_launch(void* const* d_in, const int* in_sizes, int n_in,
                              void* d_out, int out_size, void* d_ws, size_t ws_size,
                              hipStream_t stream) {
    const float* x  = (const float*)d_in[0];
    const int*   ei = (const int*)d_in[1];   // [2, E] int32
    const float* ew = (const float*)d_in[2];
    const float* W1 = (const float*)d_in[3];
    const float* b1 = (const float*)d_in[4];
    const float* W2 = (const float*)d_in[5];
    const float* b2 = (const float*)d_in[6];
    const float* Wc = (const float*)d_in[7];
    const float* bc = (const float*)d_in[8];
    float* outp = (float*)d_out;

    char* ws = (char*)d_ws;
    size_t off = 0;
    auto alloc = [&](size_t bytes) -> char* {
        char* p = ws + off;
        off = (off + bytes + 255) & ~(size_t)255;
        return p;
    };
    float*  deg    = (float*)alloc((size_t)N_NODES * 4);
    float*  dinv   = (float*)alloc((size_t)N_NODES * 4);
    float*  nrm    = (float*)alloc((size_t)N_EDGES * 4);
    __bf16* pW1    = (__bf16*)alloc((size_t)N_FEAT * N_DIMS * 2);
    __bf16* pW2    = (__bf16*)alloc((size_t)N_DIMS * N_DIMS * 2);
    __bf16* pWc    = (__bf16*)alloc((size_t)N_DIMS * 16 * 2);
    __bf16* bufA   = (__bf16*)alloc((size_t)N_NODES * N_FEAT * 2);  // packed activations
    float*  bufH   = (float*)alloc((size_t)N_NODES * N_DIMS * 4);   // gemm out
    float*  bufAgg = (float*)alloc((size_t)N_NODES * N_DIMS * 4);   // aggregated

    const int T = 256;
    const int Mtiles = N_NODES / 16;                 // 6250
    const int gemmGX = (Mtiles + 7) / 8;             // 8 waves/block

    // LDS bytes for the staged B slice: Kb * NT * 32 fragments * 32 B
    const int lds1 = (N_FEAT / 32) * 4 * 32 * 32;    // 64 KB (layer 1)
    const int lds2 = (N_DIMS / 32) * 4 * 32 * 32;    // 32 KB (layer 2)
    const int ldsc = (N_DIMS / 32) * 1 * 32 * 32;    // 8 KB  (classifier)

    // 1) normalization
    k_init_deg<<<(N_NODES + T - 1) / T, T, 0, stream>>>(deg, N_NODES);
    k_accum_deg<<<(N_EDGES + T - 1) / T, T, 0, stream>>>(ei, ew, deg, N_EDGES);
    k_dinv<<<(N_NODES + T - 1) / T, T, 0, stream>>>(deg, dinv, N_NODES);
    k_norm<<<(N_EDGES + T - 1) / T, T, 0, stream>>>(ei, ew, dinv, nrm, N_EDGES);

    // 2) pack weights (bf16 B-fragments)
    k_pack_b<<<((N_FEAT / 32) * (N_DIMS / 16) * 32 + T - 1) / T, T, 0, stream>>>(W1, pW1, N_FEAT, N_DIMS, N_DIMS);
    k_pack_b<<<((N_DIMS / 32) * (N_DIMS / 16) * 32 + T - 1) / T, T, 0, stream>>>(W2, pW2, N_DIMS, N_DIMS, N_DIMS);
    k_pack_b<<<((N_DIMS / 32) * 1 * 32 + T - 1) / T, T, 0, stream>>>(Wc, pWc, N_DIMS, N_CLSS, 16);

    // 3) layer 1: h = x @ W1 ; agg(+self loop) ; pack(relu(.+b1))
    k_pack_a<<<(Mtiles * (N_FEAT / 32) * 32 + T - 1) / T, T, 0, stream>>>(x, bufA, N_NODES, N_FEAT, nullptr);
    k_gemm_wmma<4><<<dim3(gemmGX, N_DIMS / 64), T, lds1, stream>>>(
        bufA, pW1, bufH, nullptr, Mtiles, N_FEAT / 32, N_DIMS / 16, N_DIMS, N_DIMS);
    k_agg_init<<<(N_NODES * 64 + T - 1) / T, T, 0, stream>>>(bufH, dinv, bufAgg);
    k_agg_edges<<<(int)(((long long)N_EDGES * 64 + T - 1) / T), T, 0, stream>>>(ei, nrm, bufH, bufAgg);

    // 4) layer 2: pack fuses +b1,relu ; h = h1 @ W2 ; agg
    k_pack_a<<<(Mtiles * (N_DIMS / 32) * 32 + T - 1) / T, T, 0, stream>>>(bufAgg, bufA, N_NODES, N_DIMS, b1);
    k_gemm_wmma<4><<<dim3(gemmGX, N_DIMS / 64), T, lds2, stream>>>(
        bufA, pW2, bufH, nullptr, Mtiles, N_DIMS / 32, N_DIMS / 16, N_DIMS, N_DIMS);
    k_agg_init<<<(N_NODES * 64 + T - 1) / T, T, 0, stream>>>(bufH, dinv, bufAgg);
    k_agg_edges<<<(int)(((long long)N_EDGES * 64 + T - 1) / T), T, 0, stream>>>(ei, nrm, bufH, bufAgg);

    // 5) classifier: pack fuses +b2,relu ; out = h2 @ Wc + bc (N pad 10->16)
    k_pack_a<<<(Mtiles * (N_DIMS / 32) * 32 + T - 1) / T, T, 0, stream>>>(bufAgg, bufA, N_NODES, N_DIMS, b2);
    k_gemm_wmma<1><<<dim3(gemmGX, 1), T, ldsc, stream>>>(
        bufA, pWc, outp, bc, Mtiles, N_DIMS / 32, 1, N_CLSS, N_CLSS);

    (void)in_sizes; (void)n_in; (void)out_size; (void)ws_size;
}
